// MultiHead_37984690766458
// MI455X (gfx1250) — matrix operand order repaired and checked by reading, not compiled, not run on hardware
//
#include <hip/hip_runtime.h>
#include <hip/hip_bf16.h>
#include <math.h>

// ---------------- problem constants ----------------
#define BB    8
#define TSEQ  2048
#define EMB   256
#define NH    8
#define HD    32
#define NROW  (BB*TSEQ)          // 16384 token rows
#define FFH   (4*EMB)            // 1024
#define SCALE 0.0625f            // E^-0.5 = 1/16
#define EPS   1e-5f
#define NEGBIG (-3.0e38f)

typedef __attribute__((ext_vector_type(16))) __bf16 v16bf;
typedef __attribute__((ext_vector_type(8)))  __bf16 v8bf;
typedef __attribute__((ext_vector_type(8)))  float  v8f;

__device__ __forceinline__ __bf16 f2bf(float f) {
  unsigned u = __builtin_bit_cast(unsigned, f);
  unsigned r = u + 0x7FFFu + ((u >> 16) & 1u);      // round-to-nearest-even
  unsigned short h = (unsigned short)(r >> 16);
  return __builtin_bit_cast(__bf16, h);
}

__device__ __forceinline__ v8f wmma_bf16(v16bf a, v16bf b, v8f c) {
  // D = A(16x32 bf16) x B(32x16 bf16) + C(16x16 f32)
  return __builtin_amdgcn_wmma_f32_16x16x32_bf16(false, a, false, b,
                                                 (short)0, c, false, false);
}

// A-fragment (16x32, 16-bit): lane (m = lane&15, half = lane>>4) holds
// K = half*16+e (contiguous, matches B). Two contiguous 16B loads.
__device__ __forceinline__ v16bf load_a_frag(const __bf16* rowp, int half) {
  v8bf lo = *(const v8bf*)(rowp + half*16);
  v8bf hi = *(const v8bf*)(rowp + half*16 + 8);
  v16bf a;
  for (int i = 0; i < 8; i++) { a[i] = lo[i]; a[8+i] = hi[i]; }
  return a;
}
// B-fragment (32x16): lane n holds column n; element e is K = (lane>>4)*16+e,
// i.e. 16 contiguous K values -> two contiguous 16B loads.
__device__ __forceinline__ v16bf load_b_frag_contig(const __bf16* colp) {
  v8bf lo = *(const v8bf*)colp;
  v8bf hi = *(const v8bf*)(colp + 8);
  v16bf b;
  for (int i = 0; i < 8; i++) { b[i] = lo[i]; b[8+i] = hi[i]; }
  return b;
}

// ---------------- kernel 1: weights -> bf16, TRANSPOSED [N][K] ----------------
// B^T layout makes B-fragments contiguous-K global loads (L2-resident weights;
// no LDS staging needed anywhere in the GEMMs).
__global__ __launch_bounds__(256) void k_convert(
    const float* wq, const float* wk, const float* wv,
    const float* wlin, const float* wf1, const float* wf2,
    __bf16* wqkvT, __bf16* wlinT, __bf16* wf1T, __bf16* wf2T)
{
  int i = blockIdx.x * 256 + threadIdx.x;
  int stride = gridDim.x * 256;
  // wqkvT[768][256]: row j = m*256 + h*32 + d; col e.
  for (int o = i; o < 768 * EMB; o += stride) {
    int j = o >> 8, e = o & 255;
    int m = j >> 8, hd = j & 255, h = hd >> 5, d = hd & 31;
    const float* src = (m == 0) ? wq : (m == 1) ? wk : wv;
    wqkvT[o] = f2bf(src[(h * EMB + e) * HD + d]);
  }
  // wlinT[256][256] = wlin^T
  for (int o = i; o < EMB * EMB; o += stride) {
    int j = o >> 8, e = o & 255;
    wlinT[o] = f2bf(wlin[e * EMB + j]);
  }
  // wf1T[1024][256] = wf1^T
  for (int o = i; o < FFH * EMB; o += stride) {
    int j = o >> 8, e = o & 255;
    wf1T[o] = f2bf(wf1[e * FFH + j]);
  }
  // wf2T[256][1024] = wf2^T
  for (int o = i; o < EMB * FFH; o += stride) {
    int j = o >> 10, c = o & 1023;
    wf2T[o] = f2bf(wf2[c * EMB + j]);
  }
}

// ---------------- kernel 2: LayerNorm (wave per row of 256) ----------------
__global__ __launch_bounds__(256) void k_layernorm(
    const float* __restrict__ X, const float* __restrict__ g,
    const float* __restrict__ beta, float* __restrict__ Y,
    __bf16* __restrict__ Yb)
{
  int w = threadIdx.x >> 5, lane = threadIdx.x & 31;
  int row = blockIdx.x * 8 + w;
  const float* x = X + (size_t)row * EMB;
  float v[8], s = 0.f, s2 = 0.f;
  for (int i = 0; i < 8; i++) { v[i] = x[i*32 + lane]; s += v[i]; s2 += v[i]*v[i]; }
  for (int off = 16; off; off >>= 1) {
    s  += __shfl_xor(s,  off, 32);
    s2 += __shfl_xor(s2, off, 32);
  }
  float mean = s * (1.f / EMB);
  float var  = s2 * (1.f / EMB) - mean * mean;
  float rstd = rsqrtf(var + EPS);
  for (int i = 0; i < 8; i++) {
    int e = i*32 + lane;
    float y = (v[i] - mean) * rstd * g[e] + beta[e];
    Y [(size_t)row * EMB + e] = y;
    Yb[(size_t)row * EMB + e] = f2bf(y);
  }
}

// ---------------- kernel 3: LDS-free bf16 WMMA GEMM ----------------
// C[M,N] = A[M,K] x Bt[N,K]^T. 8 waves; each wave computes a 16x64 strip
// (4 WMMAs / K-step, A-frag reused 4x), block tile = 128x64. All fragments
// are contiguous 16B global loads served by the 192MB L2. No LDS, no barriers.
// KC/NC compile-time -> full address strength reduction + clause batching.
// MODE 0: +bias, optional relu / f32 residual, f32 and/or bf16 out.
// MODE 1: QKV scatter epilogue (q,k: [bh][t][d] bf16; v transposed [bh][d][t]).
template<int MODE, int KC, int NC>
__global__ __launch_bounds__(256) void k_gemm(
    const __bf16* __restrict__ A, const __bf16* __restrict__ Bt,
    const float* __restrict__ bias, int relu, const float* __restrict__ res,
    float* __restrict__ outF, __bf16* __restrict__ outB,
    __bf16* __restrict__ qO, __bf16* __restrict__ kO, __bf16* __restrict__ vTO)
{
  int tid = threadIdx.x;
  int w = tid >> 5, lane = tid & 31;
  int half = lane >> 4, ln = lane & 15;
  int bm = blockIdx.x * 128, bn = blockIdx.y * 64;

  const __bf16* arow = A + (size_t)(bm + w * 16 + ln) * KC;
  const __bf16* brow[4];
  for (int ct = 0; ct < 4; ct++)
    brow[ct] = Bt + (size_t)(bn + ct * 16 + ln) * KC + half * 16;

  v8f c[4];
  for (int ct = 0; ct < 4; ct++) c[ct] = (v8f){0,0,0,0,0,0,0,0};

#pragma unroll 4
  for (int k0 = 0; k0 < KC; k0 += 32) {
    v16bf a = load_a_frag(arow + k0, half);
    for (int ct = 0; ct < 4; ct++) {
      v16bf b = load_b_frag_contig(brow[ct] + k0);
      c[ct] = wmma_bf16(a, b, c[ct]);
    }
  }

  for (int ct = 0; ct < 4; ct++) {
    for (int r = 0; r < 8; r++) {
      int row = bm + w * 16 + r + half * 8;
      int col = bn + ct * 16 + ln;
      float val = c[ct][r];
      if (MODE == 0) {
        if (bias) val += bias[col];
        if (relu) val = fmaxf(val, 0.f);
        if (res)  val += res[(size_t)row * NC + col];
        if (outF) outF[(size_t)row * NC + col] = val;
        if (outB) outB[(size_t)row * NC + col] = f2bf(val);
      } else {
        int m = col >> 8, hd = col & 255, h = hd >> 5, d = hd & 31;
        int b = row >> 11, t = row & (TSEQ - 1);
        size_t bh = (size_t)(b * NH + h);
        if (m == 0)      qO [(bh * TSEQ + t) * HD + d] = f2bf(val);
        else if (m == 1) kO [(bh * TSEQ + t) * HD + d] = f2bf(val);
        else             vTO[(bh * HD + d) * TSEQ + t] = f2bf(val);
      }
    }
  }
}

// ---------------- kernel 4: per-column softmax stats (branchless) ----------------
// softmax is over the QUERY axis: for key column s, normalize over t in [s,T).
// One wave owns a 16-column tile; each lane has a fixed column. Masked scores
// are selected to -3e38 (v_cndmask, no EXEC branches); the diagonal guarantees
// every column sees >=1 valid element in its first tile; lanes 8-15 see none there and hold m = -3e38,
// z = 8 until a finite max arrives (next tile or half-lane merge), where exp(-3e38 - mn) flushes to 0.
__global__ __launch_bounds__(256) void k_softmax_stats(
    const __bf16* __restrict__ q, const __bf16* __restrict__ k,
    float* __restrict__ sm, float* __restrict__ sz)
{
  int bh = blockIdx.x;
  int w = threadIdx.x >> 5, lane = threadIdx.x & 31;
  int half = lane >> 4, ln = lane & 15;
  int sTile = blockIdx.y * 8 + w;
  int s0 = sTile * 16;
  v16bf kb = load_b_frag_contig(
      k + ((size_t)bh * TSEQ + s0 + ln) * HD + half * 16);
  float m = NEGBIG, z = 0.f;
  int s = s0 + ln;
  const __bf16* qbase = q + ((size_t)bh * TSEQ + ln) * HD;
  for (int tt = sTile; tt < TSEQ / 16; tt++) {
    int t0 = tt * 16;
    if (tt + 2 < TSEQ / 16)
      __builtin_prefetch(qbase + (size_t)(t0 + 32) * HD, 0, 0);
    v16bf qa = load_a_frag(qbase + (size_t)t0 * HD, half);
    v8f c = {0,0,0,0,0,0,0,0};
    c = wmma_bf16(qa, kb, c);
    float xs[8], mt = NEGBIG;
    for (int r = 0; r < 8; r++) {
      int t = t0 + r + half * 8;
      xs[r] = (t >= s) ? c[r] * SCALE : NEGBIG;   // cndmask, branchless
      mt = fmaxf(mt, xs[r]);
    }
    float mn = fmaxf(m, mt);
    float zt = 0.f;
    for (int r = 0; r < 8; r++) zt += __expf(xs[r] - mn);
    z = z * __expf(m - mn) + zt;
    m = mn;
  }
  { // merge the two half-lanes that share a column
    float m2 = __shfl_xor(m, 16, 32);
    float z2 = __shfl_xor(z, 16, 32);
    float mn = fmaxf(m, m2);
    z = z * __expf(m - mn) + z2 * __expf(m2 - mn);
    m = mn;
  }
  if (half == 0) {
    sm[(size_t)bh * TSEQ + s] = m;
    sz[(size_t)bh * TSEQ + s] = z;
  }
}

// ---------------- kernel 5: heads = p @ V, fused xn residual ----------------
__global__ __launch_bounds__(256) void k_attn_heads(
    const __bf16* __restrict__ q, const __bf16* __restrict__ k,
    const __bf16* __restrict__ vT, const float* __restrict__ sm,
    const float* __restrict__ sz, const float* __restrict__ xn,
    __bf16* __restrict__ xhb)
{
  __shared__ __bf16 pbuf[8][16 * 32];     // per-wave p staging (C->A relayout)
  int bh = blockIdx.x;
  int b = bh >> 3, h = bh & 7;
  int w = threadIdx.x >> 5, lane = threadIdx.x & 31;
  int half = lane >> 4, ln = lane & 15;
  int tt = blockIdx.y * 8 + w;
  int t0 = tt * 16;
  __bf16* pb = &pbuf[w][0];

  v16bf qa = load_a_frag(q + ((size_t)bh * TSEQ + t0 + ln) * HD, half);
  v8f c0 = {0,0,0,0,0,0,0,0}, c1 = {0,0,0,0,0,0,0,0};

  int scMax = tt >> 1;                     // last 32-wide s-chunk touching t0+15
  for (int sc = 0; sc <= scMax; sc++) {
    if (sc + 1 <= scMax) {                 // prefetch next chunk of k and vT
      __builtin_prefetch(k + ((size_t)bh * TSEQ + (sc + 1) * 32 + ln) * HD, 0, 0);
      __builtin_prefetch(vT + ((size_t)bh * HD + ln) * TSEQ + (sc + 1) * 32, 0, 0);
    }
    for (int sub = 0; sub < 2; sub++) {
      int s0 = sc * 32 + sub * 16;
      if (s0 <= t0 + 15) {                 // wave-uniform branch
        v16bf kb = load_b_frag_contig(
            k + ((size_t)bh * TSEQ + s0 + ln) * HD + half * 16);
        v8f sf = {0,0,0,0,0,0,0,0};
        sf = wmma_bf16(qa, kb, sf);
        int s = s0 + ln;                   // fixed column per lane
        float mcol = sm[(size_t)bh * TSEQ + s];
        float invz = 1.f / sz[(size_t)bh * TSEQ + s];
        for (int r = 0; r < 8; r++) {
          int t = t0 + r + half * 8;
          float e = __expf(sf[r] * SCALE - mcol) * invz;
          float p = (s <= t) ? e : 0.f;    // cndmask, branchless
          pb[(r + half * 8) * 32 + sub * 16 + ln] = f2bf(p);
        }
      } else {                             // fully masked sub-tile
        for (int r = 0; r < 8; r++)
          pb[(r + half * 8) * 32 + sub * 16 + ln] = f2bf(0.f);
      }
    }
    // per-wave LDS RAW: LDS is in-order per wave; fence compiler + counters
    asm volatile("s_wait_dscnt 0" ::: "memory");
    v16bf pa = load_a_frag(pb + ln * 32, half);
    v16bf vb0 = load_b_frag_contig(
        vT + ((size_t)bh * HD + ln) * TSEQ + sc * 32 + half * 16);
    v16bf vb1 = load_b_frag_contig(
        vT + ((size_t)bh * HD + 16 + ln) * TSEQ + sc * 32 + half * 16);
    c0 = wmma_bf16(pa, vb0, c0);
    c1 = wmma_bf16(pa, vb1, c1);
  }
  for (int ct = 0; ct < 2; ct++) {
    v8f& c = ct ? c1 : c0;
    for (int r = 0; r < 8; r++) {
      int t = t0 + r + half * 8;
      int d = ct * 16 + ln;
      size_t idx = ((size_t)b * TSEQ + t) * EMB + h * HD + d;
      xhb[idx] = f2bf(xn[idx] + c[r]);     // residual around attention (on xn)
    }
  }
}

// ---------------- host orchestration ----------------
extern "C" void kernel_launch(void* const* d_in, const int* in_sizes, int n_in,
                              void* d_out, int out_size, void* d_ws, size_t ws_size,
                              hipStream_t stream) {
  const float* x     = (const float*)d_in[0];
  const float* wq    = (const float*)d_in[1];
  const float* wk    = (const float*)d_in[2];
  const float* wv    = (const float*)d_in[3];
  const float* wlin  = (const float*)d_in[4];
  const float* blin  = (const float*)d_in[5];
  const float* g1    = (const float*)d_in[6];
  const float* beta1 = (const float*)d_in[7];
  const float* g2    = (const float*)d_in[8];
  const float* beta2 = (const float*)d_in[9];
  const float* wf1   = (const float*)d_in[10];
  const float* bf1   = (const float*)d_in[11];
  const float* wf2   = (const float*)d_in[12];
  const float* bf2   = (const float*)d_in[13];
  float* out = (float*)d_out;

  char* ws = (char*)d_ws;
  size_t off = 0;
  auto carve = [&](size_t bytes) -> char* {
    char* p = ws + off;
    off = (off + bytes + 255) & ~(size_t)255;
    return p;
  };
  float*  xn    = (float*) carve((size_t)NROW * EMB * 4);
  __bf16* xnb   = (__bf16*)carve((size_t)NROW * EMB * 2);
  __bf16* qb    = (__bf16*)carve((size_t)BB * NH * TSEQ * HD * 2);
  __bf16* kb    = (__bf16*)carve((size_t)BB * NH * TSEQ * HD * 2);
  __bf16* vTb   = (__bf16*)carve((size_t)BB * NH * TSEQ * HD * 2);
  float*  smS   = (float*) carve((size_t)BB * NH * TSEQ * 4);
  float*  szS   = (float*) carve((size_t)BB * NH * TSEQ * 4);
  __bf16* xhb   = (__bf16*)carve((size_t)NROW * EMB * 2);
  float*  xl    = (float*) carve((size_t)NROW * EMB * 4);
  float*  x2    = (float*) carve((size_t)NROW * EMB * 4);
  __bf16* x2b   = (__bf16*)carve((size_t)NROW * EMB * 2);
  __bf16* h1b   = (__bf16*)carve((size_t)NROW * FFH * 2);
  __bf16* wqkvT = (__bf16*)carve((size_t)768 * EMB * 2);
  __bf16* wlinT = (__bf16*)carve((size_t)EMB * EMB * 2);
  __bf16* wf1T  = (__bf16*)carve((size_t)FFH * EMB * 2);
  __bf16* wf2T  = (__bf16*)carve((size_t)EMB * FFH * 2);
  (void)ws_size; (void)in_sizes; (void)n_in; (void)out_size;

  // 1. weights -> bf16, transposed [N][K]
  k_convert<<<1024, 256, 0, stream>>>(wq, wk, wv, wlin, wf1, wf2,
                                      wqkvT, wlinT, wf1T, wf2T);
  // 2. LN1
  k_layernorm<<<NROW / 8, 256, 0, stream>>>(x, g1, beta1, xn, xnb);
  // 3. fused QKV projection (M=16384, N=768, K=256)
  k_gemm<1, EMB, 768><<<dim3(NROW / 128, 768 / 64), 256, 0, stream>>>(
      xnb, wqkvT, nullptr, 0, nullptr, nullptr, nullptr, qb, kb, vTb);
  // 4. column softmax stats
  k_softmax_stats<<<dim3(BB * NH, TSEQ / 16 / 8), 256, 0, stream>>>(
      qb, kb, smS, szS);
  // 5. heads + residual -> xh (bf16)
  k_attn_heads<<<dim3(BB * NH, TSEQ / 16 / 8), 256, 0, stream>>>(
      qb, kb, vTb, smS, szS, xn, xhb);
  // 6. projection: xl = xh @ w_lin + b_lin
  k_gemm<0, EMB, EMB><<<dim3(NROW / 128, EMB / 64), 256, 0, stream>>>(
      xhb, wlinT, blin, 0, nullptr, xl, nullptr, nullptr, nullptr, nullptr);
  // 7. LN2
  k_layernorm<<<NROW / 8, 256, 0, stream>>>(xl, g2, beta2, x2, x2b);
  // 8. FFN up: h1 = relu(x2 @ w_f1 + b_f1)  (bf16 out)
  k_gemm<0, EMB, FFH><<<dim3(NROW / 128, FFH / 64), 256, 0, stream>>>(
      x2b, wf1T, bf1, 1, nullptr, nullptr, h1b, nullptr, nullptr, nullptr);
  // 9. FFN down + residual: out = x2 + h1 @ w_f2 + b_f2
  k_gemm<0, FFH, EMB><<<dim3(NROW / 128, EMB / 64), 256, 0, stream>>>(
      h1b, wf2T, bf2, 0, x2, out, nullptr, nullptr, nullptr, nullptr);
}
